// CapsuleLayer_2808908612300
// MI455X (gfx1250) — matrix operand order repaired and checked by reading, not compiled, not run on hardware
//
#include <hip/hip_runtime.h>

// ---------------------------------------------------------------------------
// Capsule routing on MI455X (gfx1250).
// pred ([B,P,N,T] = 377 MB) is never materialized. Routing coefficients are
// batch-independent (bias updates are batch-mean broadcasts), so the whole
// loop is 5 bf16-WMMA GEMMs + tiny elementwise kernels, all L2-resident.
//   B=512, P=1152, N=10, T=16, D=8, K1=P*D=9216, NT=N*T=160, NUM_ITER=3
// ---------------------------------------------------------------------------

#define BB   512
#define PP   1152
#define NN   10
#define TT   16
#define DD   8
#define K1   (PP*DD)      // 9216
#define NT   (NN*TT)      // 160
#define KSLICES 8         // split-K factor for the s-GEMM (occupancy)
#define SLICE_LEN (K1/KSLICES)   // 1152 bf16 elements = 576 dwords

typedef __bf16 v16bf __attribute__((ext_vector_type(16)));
typedef float  v8f   __attribute__((ext_vector_type(8)));
typedef unsigned int v8u __attribute__((ext_vector_type(8)));

union FragU { v8u u; v16bf b; };

static __device__ __forceinline__ unsigned short f32_to_bf16(float f) {
    unsigned int u = __float_as_uint(f);
    u += 0x7FFFu + ((u >> 16) & 1u);   // round-to-nearest-even
    return (unsigned short)(u >> 16);
}

// ---------------------------------------------------------------------------
__global__ __launch_bounds__(256) void k_zero(float* __restrict__ bias) {
    int i = blockIdx.x * 256 + threadIdx.x;     // grid covers PP*NN exactly
    bias[i] = 0.0f;
}

// Convert x (f32 [B][P*D]) to bf16 in row-major and transposed layouts.
__global__ __launch_bounds__(256) void k_convert_x(const float* __restrict__ x,
                                                   unsigned short* __restrict__ xbf,
                                                   unsigned short* __restrict__ xT) {
    int idx = blockIdx.x * 256 + threadIdx.x;   // grid covers B*K1 exactly
    int b  = idx / K1;
    int pd = idx - b * K1;
    unsigned short h = f32_to_bf16(x[idx]);
    xbf[idx] = h;
    xT[(size_t)pd * BB + b] = h;
}

// Softmax over N per p (coupling coefficients are batch-independent).
__global__ __launch_bounds__(256) void k_softmax(const float* __restrict__ bias,
                                                 float* __restrict__ c) {
    int p = blockIdx.x * 256 + threadIdx.x;
    if (p >= PP) return;
    float m = -3.4e38f;
    float e[NN];
    for (int n = 0; n < NN; ++n) m = fmaxf(m, bias[p * NN + n]);
    float sum = 0.0f;
    for (int n = 0; n < NN; ++n) { e[n] = __expf(bias[p * NN + n] - m); sum += e[n]; }
    float inv = 1.0f / sum;
    for (int n = 0; n < NN; ++n) c[p * NN + n] = e[n] * inv;
}

// WcT[nt][pd] = bf16( c[p,n] * W[p,n,t,d] )   (K-contiguous B operand)
__global__ __launch_bounds__(256) void k_scaleW(const float* __restrict__ W,
                                                const float* __restrict__ c,
                                                unsigned short* __restrict__ wcT) {
    int idx = blockIdx.x * 256 + threadIdx.x;   // grid covers NT*K1 exactly
    int nt = idx / K1;
    int pd = idx - nt * K1;
    int n = nt >> 4, t = nt & 15;
    int p = pd >> 3, d = pd & 7;
    float v = c[p * NN + n] * W[(((size_t)p * NN + n) * TT + t) * DD + d];
    wcT[idx] = f32_to_bf16(v);
}

// ---------------------------------------------------------------------------
// bf16 WMMA GEMM with split-K. All K indexing is in DWORD units (2 bf16) off
// pre-offset row pointers, so fragment accesses are constant offsets from one
// induction variable -> compiler fuses them into global_load_b128 + immediate
// offsets (no per-iteration address VALU, no parity proofs needed).
//   A rows / Bt rows: K-contiguous bf16, row length rowLen elements.
//   blockIdx.y = K-slice (sliceLenD dwords); exact grids -> EXEC all-ones.
__global__ __launch_bounds__(256) void k_gemm_bf16(const unsigned short* __restrict__ A,
                                                   const unsigned short* __restrict__ Bt,
                                                   float* __restrict__ C,
                                                   int rowLen, int sliceLenD,
                                                   long long cSliceStride) {
    int lane = threadIdx.x & 31;
    int wave = threadIdx.x >> 5;
    int tile = blockIdx.x * 8 + wave;
    int mt = tile / NN;               // C row-tile
    int nt = tile - mt * NN;          // C col-tile (0..9)
    int m0 = mt * 16, n0 = nt * 16;
    float* Cs = C + (long long)blockIdx.y * cSliceStride;

    int am = lane & 15;               // row (A) / col (B) handled by this lane
    int hi = lane >> 4;               // upper half-wave

    int kdBase = blockIdx.y * sliceLenD;   // slice base, in dwords
    // A frag dword idx (pre-offset +hi*4): kd + (i&3) + (i>=4 ? 8 : 0)
    const unsigned int* aD = (const unsigned int*)(A + (size_t)(m0 + am) * rowLen)
                             + kdBase + hi * 4;
    // B frag dword idx (pre-offset +hi*4): kd + (j&3) + (j>=4 ? 8 : 0)
    const unsigned int* bD = (const unsigned int*)(Bt + (size_t)(n0 + am) * rowLen)
                             + kdBase + hi * 4;

    v8f acc = {};
#pragma unroll 2
    for (int kd = 0; kd < sliceLenD; kd += 16) {   // 16 dwords = 32 bf16 = one WMMA K
        FragU Af, Bf;
#pragma unroll
        for (int i = 0; i < 4; ++i) {
            Af.u[i]     = aD[kd + i];          // 16B contiguous
            Af.u[4 + i] = aD[kd + 8 + i];      // 16B contiguous
        }
#pragma unroll
        for (int j = 0; j < 4; ++j) {
            Bf.u[j]     = bD[kd + j];          // 16B contiguous
            Bf.u[4 + j] = bD[kd + 8 + j];      // 16B contiguous
        }
        acc = __builtin_amdgcn_wmma_f32_16x16x32_bf16(
                  false, Af.b, false, Bf.b, (short)0, acc, false, false);
    }
    // C/D layout: vgpr r -> row m0 + r + hi*8, col n0 + (lane&15)
#pragma unroll
    for (int r = 0; r < 8; ++r)
        Cs[(size_t)(m0 + r + hi * 8) * NT + (n0 + am)] = acc[r];
}

// Reduce split-K partials: s[i] = sum_slice Cpart[slice][i]
__global__ __launch_bounds__(256) void k_reduce_k(const float* __restrict__ Cpart,
                                                  float* __restrict__ s) {
    int idx = blockIdx.x * 256 + threadIdx.x;   // grid covers B*NT exactly
    float acc = 0.0f;
#pragma unroll
    for (int sl = 0; sl < KSLICES; ++sl)
        acc += Cpart[(size_t)sl * (BB * NT) + idx];
    s[idx] = acc;
}

// ---------------------------------------------------------------------------
// squash: v = |s|^2 * s / (1 + |s|^2 (|s| + 1e-9)); emit vT bf16 for the
// agreement GEMM and (last iter) the final f32 output [B][N][T][1].
__global__ __launch_bounds__(256) void k_squash(const float* __restrict__ s,
                                                unsigned short* __restrict__ vT,
                                                float* __restrict__ out,
                                                int write_out) {
    int idx = blockIdx.x * 256 + threadIdx.x;   // grid covers B*N exactly
    int b = idx / NN;
    int n = idx - b * NN;
    const float* sp = s + (size_t)b * NT + n * TT;
    float tmp[TT];
    float sq = 0.0f;
#pragma unroll
    for (int t = 0; t < TT; ++t) { tmp[t] = sp[t]; sq += tmp[t] * tmp[t]; }
    float norm  = sqrtf(sq);
    float scale = sq / (1.0f + sq * (norm + 1e-9f));
#pragma unroll
    for (int t = 0; t < TT; ++t) {
        float vv = scale * tmp[t];
        vT[(size_t)(n * TT + t) * BB + b] = f32_to_bf16(vv);
        if (write_out) out[(size_t)b * NT + n * TT + t] = vv;
    }
}

// bias[p,n] += (1/B) * sum_{t,d} W[p,n,t,d] * M[(p,d),(n,t)]
__global__ __launch_bounds__(256) void k_bias_update(const float* __restrict__ W,
                                                     const float* __restrict__ M,
                                                     float* __restrict__ bias) {
    int idx = blockIdx.x * 256 + threadIdx.x;   // grid covers PP*NN exactly
    int p = idx / NN;
    int n = idx - p * NN;
    const float* wp = W + (((size_t)p * NN + n) * TT) * DD;
    float acc = 0.0f;
#pragma unroll
    for (int t = 0; t < TT; ++t)
#pragma unroll
        for (int d = 0; d < DD; ++d)
            acc += wp[t * DD + d] * M[(size_t)(p * DD + d) * NT + n * TT + t];
    bias[idx] += acc * (1.0f / (float)BB);
}

// ---------------------------------------------------------------------------
extern "C" void kernel_launch(void* const* d_in, const int* in_sizes, int n_in,
                              void* d_out, int out_size, void* d_ws, size_t ws_size,
                              hipStream_t stream) {
    const float* x = (const float*)d_in[0];   // [512,1152,8]
    const float* W = (const float*)d_in[1];   // [1152,10,16,8]
    float* out = (float*)d_out;               // [512,10,16,1]

    char* ws = (char*)d_ws;
    // 256B-aligned workspace carve-up (~29.5 MB total)
    unsigned short* xbf   = (unsigned short*)(ws);                      //  9,437,184 B
    unsigned short* xT    = (unsigned short*)(ws + 9437184);            //  9,437,184 B
    unsigned short* wcT   = (unsigned short*)(ws + 18874368);           //  2,949,120 B
    float*          sbuf  = (float*)(ws + 21823488);                    //    327,680 B
    unsigned short* vT    = (unsigned short*)(ws + 22151168);           //    163,840 B
    float*          Mbuf  = (float*)(ws + 22315008);                    //  5,898,240 B
    float*          bias  = (float*)(ws + 28213248);                    //     46,080 B
    float*          coef  = (float*)(ws + 28259328);                    //     46,080 B
    float*          cpart = (float*)(ws + 28305408);                    //  2,621,440 B

    k_zero     <<<(PP*NN)/256, 256, 0, stream>>>(bias);                 // 45 blocks
    k_convert_x<<<(BB*K1)/256, 256, 0, stream>>>(x, xbf, xT);           // 18432 blocks

    for (int it = 0; it < 3; ++it) {
        k_softmax<<<(PP + 255)/256, 256, 0, stream>>>(bias, coef);
        k_scaleW <<<(NT*K1)/256,    256, 0, stream>>>(W, coef, wcT);    // 5760 blocks
        // s[512x160] = x_bf[512x9216] * WcT^T, split-K x8:
        //   grid (40 tile-blocks, 8 slices) = 2560 waves
        k_gemm_bf16<<<dim3(40, KSLICES), 256, 0, stream>>>(
            xbf, wcT, cpart, K1, SLICE_LEN / 2, (long long)BB * NT);
        k_reduce_k <<<(BB*NT)/256,  256, 0, stream>>>(cpart, sbuf);     // 320 blocks
        k_squash   <<<(BB*NN)/256,  256, 0, stream>>>(sbuf, vT, out, (it == 2) ? 1 : 0);
        if (it < 2) {
            // M[9216x160] = xT[9216x512] * vT^T : 5760 tiles -> 720 blocks, no split
            k_gemm_bf16<<<dim3(720, 1), 256, 0, stream>>>(
                xT, vT, Mbuf, BB, BB / 2, 0LL);
            k_bias_update<<<(PP*NN)/256, 256, 0, stream>>>(W, Mbuf, bias);
        }
    }
}
